// DynamicEdgeProcessor_25099788878454
// MI455X (gfx1250) — matrix operand order repaired and checked by reading, not compiled, not run on hardware
//
#include <hip/hip_runtime.h>

// MI455X / gfx1250, wave32. Matrix math via v_wmma_f32_16x16x32_f16; distance
// row staged into LDS via the Tensor Data Mover (tensor_load_to_lds +
// s_wait_tensorcnt). Constant WMMA operand fragments are pre-swizzled once
// into workspace so hot kernels fetch them with contiguous 32B vector loads.
//
// Roofline: edge_features output = 67MB -> ~3us floor at 23.3 TB/s. Algebraic
// factoring (P_i/P_j precompute + folding dist-encoder layer 2 into uW1)
// cuts FLOPs ~20G -> ~2G, so WMMA keeps the hot kernel bandwidth-bound.

typedef __attribute__((ext_vector_type(16))) _Float16 v16h;
typedef __attribute__((ext_vector_type(8)))  float    v8f;
typedef __attribute__((ext_vector_type(4)))  unsigned u32x4;
typedef __attribute__((ext_vector_type(8)))  int      i32x8;
typedef __attribute__((ext_vector_type(4)))  int      i32x4;

#define Bb 16
#define Nn 256
#define Hh 128
#define Ee 16
#define LN_EPS 1e-5f

// Workspace layout (bytes), all 32B aligned.
#define OFF_P    0u         // P[node][64] f32: 4096*64*4   = 1048576
#define OFF_POOL 1048576u   // pooled sums f32: 4096*16*4   = 262144
#define OFF_CC   1310720u   // colconst f32[32]             = 128
#define OFF_BM   1310848u   // bm frags f16 [2][32][16]     = 2048
#define OFF_B2   1312896u   // uW2 frag f16 [32][16]        = 1024
#define OFF_UW1F 1313920u   // kP B frags f16 [16][32][16]  = 16384
#define OFF_NW1F 1330304u   // kmsg L1 frags f16 [4][32][16]= 4096
#define OFF_NW2F 1334400u   // kmsg L2 frags f16 [16][32][16]=16384

// A-matrix f16 16x32 layout: lane (hf=lane/16, ln=lane%16) holds row M=ln;
// VGPR pair p holds K = (p<4?0:16) + hf*8 + 2*(p&3) and K+1 (ISA 7.12.2).
__device__ __forceinline__ int kb(int p, int hf) {
  return ((p & 4) << 2) + (hf << 3) + ((p & 3) << 1);
}

__device__ __forceinline__ v8f wmma_f16(v16h a, v16h b, v8f c) {
  return __builtin_amdgcn_wmma_f32_16x16x32_f16(false, a, false, b, (short)0, c,
                                                false, false);
}

union HU { unsigned u; _Float16 h[2]; };

// ---- one-time constant folding + operand-fragment pre-swizzle --------------
__global__ void kconst_kernel(const float* __restrict__ dW2,
                              const float* __restrict__ db2,
                              const float* __restrict__ uW1,
                              const float* __restrict__ ub1,
                              const float* __restrict__ uW2,
                              const float* __restrict__ nW1,
                              const float* __restrict__ nW2,
                              float* __restrict__ colconst,
                              _Float16* __restrict__ bmf,
                              _Float16* __restrict__ b2f,
                              _Float16* __restrict__ uw1f,
                              _Float16* __restrict__ nw1f,
                              _Float16* __restrict__ nw2f) {
  int lane = threadIdx.x, hf = lane >> 4, ln = lane & 15;
  // colconst = ub1 + db2 @ uW1[256:272]
  float cc = ub1[lane];
  for (int e = 0; e < Ee; ++e) cc += db2[e] * uW1[(2 * Hh + e) * 32 + lane];
  colconst[lane] = cc;
  // bm = (dW2 @ uW1[256:272]) as two B frags, K padded 16->32 (q>=8 zero)
  for (int t = 0; t < 2; ++t)
    for (int q = 0; q < 16; ++q) {
      float s = 0.f;
      if (q < 8)
        for (int e = 0; e < Ee; ++e)
          s += dW2[(hf * 8 + q) * Ee + e] * uW1[(2 * Hh + e) * 32 + t * 16 + ln];
      bmf[(t * 32 + lane) * 16 + q] = (_Float16)s;
    }
  // uW2 (32x16) B frag
  for (int q = 0; q < 16; ++q)
    b2f[lane * 16 + q] = (_Float16)uW2[((q & 8) * 2 + hf * 8 + (q & 7)) * Ee + ln];
  // kP B frags from uW1[0:256]
  for (int kc = 0; kc < 4; ++kc)
    for (int t = 0; t < 4; ++t)
      for (int q = 0; q < 16; ++q) {
        int k = (t >> 1) * Hh + kc * 32 + (q & 8) * 2 + hf * 8 + (q & 7);
        uw1f[((kc * 4 + t) * 32 + lane) * 16 + q] =
            (_Float16)uW1[k * 32 + (t & 1) * 16 + ln];
      }
  // kmsg layer-1 frags (K padded 16->32) and layer-2 frags
  for (int t = 0; t < 4; ++t)
    for (int q = 0; q < 16; ++q)
      nw1f[(t * 32 + lane) * 16 + q] =
          (_Float16)(q < 8 ? nW1[(hf * 8 + q) * 64 + t * 16 + ln] : 0.f);
  for (int t2 = 0; t2 < 8; ++t2)
    for (int kc = 0; kc < 2; ++kc)
      for (int q = 0; q < 16; ++q)
        nw2f[((t2 * 2 + kc) * 32 + lane) * 16 + q] =
            (_Float16)nW2[(kc * 32 + (q & 8) * 2 + hf * 8 + (q & 7)) * Hh + t2 * 16 + ln];
}

// ---- P precompute: P[node][0:32]=emb@uW1[0:128], [32:64]=emb@uW1[128:256] --
__global__ void kP_kernel(const float* __restrict__ emb,
                          const _Float16* __restrict__ uw1f,
                          float* __restrict__ P) {
  int lane = threadIdx.x & 31, hf = lane >> 4, ln = lane & 15;
  int node0 = blockIdx.x * 16;
  const float4* e4 = (const float4*)(emb + (size_t)(node0 + ln) * Hh);
  v16h a[4];
#pragma unroll
  for (int kc = 0; kc < 4; ++kc) {
    float4 lo0 = e4[(kc * 32 + hf * 8) >> 2];
    float4 lo1 = e4[((kc * 32 + hf * 8) >> 2) + 1];
    float4 hi0 = e4[(kc * 32 + 16 + hf * 8) >> 2];
    float4 hi1 = e4[((kc * 32 + 16 + hf * 8) >> 2) + 1];
    a[kc][0] = (_Float16)lo0.x;  a[kc][1] = (_Float16)lo0.y;
    a[kc][2] = (_Float16)lo0.z;  a[kc][3] = (_Float16)lo0.w;
    a[kc][4] = (_Float16)lo1.x;  a[kc][5] = (_Float16)lo1.y;
    a[kc][6] = (_Float16)lo1.z;  a[kc][7] = (_Float16)lo1.w;
    a[kc][8] = (_Float16)hi0.x;  a[kc][9] = (_Float16)hi0.y;
    a[kc][10] = (_Float16)hi0.z; a[kc][11] = (_Float16)hi0.w;
    a[kc][12] = (_Float16)hi1.x; a[kc][13] = (_Float16)hi1.y;
    a[kc][14] = (_Float16)hi1.z; a[kc][15] = (_Float16)hi1.w;
  }
#pragma unroll
  for (int t = 0; t < 4; ++t) {
    v8f c = {};
#pragma unroll
    for (int kc = 0; kc < 4; ++kc) {
      v16h bfr = *(const v16h*)(uw1f + ((kc * 4 + t) * 32 + lane) * 16);
      c = wmma_f16(a[kc], bfr, c);
    }
#pragma unroll
    for (int r = 0; r < 8; ++r)
      P[(node0 + r + 8 * hf) * 64 + (t >> 1) * 32 + (t & 1) * 16 + ln] = c[r];
  }
}

// ---- hot kernel: one block per (b,i); 4 waves split 16 j-tiles -------------
__global__ void __launch_bounds__(128)
kedge_kernel(const float* __restrict__ dist, const int* __restrict__ visited,
             const float* __restrict__ dW1, const float* __restrict__ db1,
             const float* __restrict__ lng, const float* __restrict__ lnb,
             const float* __restrict__ ub2, const float* __restrict__ P,
             const _Float16* __restrict__ bmf, const _Float16* __restrict__ b2f,
             const float* __restrict__ colconst,
             float* __restrict__ ef, float* __restrict__ pooled) {
  __shared__ _Float16 hbuf[4 * 16 * 32];  // per-wave transpose staging
  __shared__ float pbuf[4 * 16];
  __shared__ float dbuf[Nn];              // TDM-staged distance row
  int lane = threadIdx.x & 31, wv = threadIdx.x >> 5, hf = lane >> 4,
      ln = lane & 15;
  int b = blockIdx.x >> 8, i = blockIdx.x & 255;
  _Float16* lw = hbuf + wv * 16 * 32;
  const float* drow = dist + (size_t)(b * Nn + i) * Nn;

  // ---- stage dist row (1KB) into LDS via Tensor Data Mover ----
#if __has_builtin(__builtin_amdgcn_tensor_load_to_lds)
  if (wv == 0) {
    unsigned long long ga = (unsigned long long)(uintptr_t)drow;
    unsigned lds_off = (unsigned)(uintptr_t)&dbuf[0];
    // D# group0: count=1; lds_addr; global_addr[56:0]; type=2 (bits 127:126)
    u32x4 g0 = {1u, lds_off, (unsigned)ga,
                (unsigned)((ga >> 32) & 0x1FFFFFFull) | 0x80000000u};
    // D# group1: data_size=4B; tensor_dim0=256, tensor_dim1=1;
    //            tile_dim0=256, tile_dim1=1; tensor_dim0_stride=256
    i32x8 g1 = {0x00020000, 0x01000000, 0x00010000, 0x01000000, 1, 256, 0, 0};
    i32x4 g2 = {0, 0, 0, 0}, g3 = {0, 0, 0, 0};
#if defined(__clang_major__) && __clang_major__ >= 23
    i32x8 g4 = {0, 0, 0, 0, 0, 0, 0, 0};
    __builtin_amdgcn_tensor_load_to_lds(g0, g1, g2, g3, g4, 0);
#else
    __builtin_amdgcn_tensor_load_to_lds(g0, g1, g2, g3, 0);
#endif
#if __has_builtin(__builtin_amdgcn_s_wait_tensorcnt)
    __builtin_amdgcn_s_wait_tensorcnt(0);
#else
    asm volatile("s_wait_tensorcnt 0x0" ::: "memory");
#endif
  }
#else
  for (int t = threadIdx.x; t < Nn; t += 128) dbuf[t] = drow[t];
#endif
  __syncthreads();

  // loop-invariant fragments / scalars (all contiguous vector loads)
  float dw0[4], dw1[4], dv0[4], dv1[4];
#pragma unroll
  for (int p = 0; p < 4; ++p) {
    int k = hf * 8 + 2 * p;
    dw0[p] = dW1[k]; dw1[p] = dW1[k + 1];
    dv0[p] = db1[k]; dv1[p] = db1[k + 1];
  }
  v16h bm0 = *(const v16h*)(bmf + (0 * 32 + lane) * 16);
  v16h bm1 = *(const v16h*)(bmf + (1 * 32 + lane) * 16);
  v16h b2  = *(const v16h*)(b2f + lane * 16);
  float g0v = lng[ln], g1v = lng[16 + ln], be0 = lnb[ln], be1 = lnb[16 + ln];
  float cc0 = colconst[ln], cc1 = colconst[16 + ln];
  const float* Pi = P + (b * Nn + i) * 64;
  float pi0 = Pi[ln], pi1 = Pi[16 + ln];
  float scale = visited[b * Nn + i] ? 0.5f : 1.0f;
  float u2 = ub2[ln];
  float poolacc = 0.f;

  for (int jt = wv; jt < 16; jt += 4) {
    int j0 = jt * 16;
    if (jt + 4 < 16)  // prefetch next tile's P_j columns
      __builtin_prefetch(P + (b * Nn + j0 + 64) * 64 + 32, 0, 1);
    float d = dbuf[j0 + ln];
    // dist-encoder hidden vec as WMMA A rows (K=16 padded to 32)
    v16h a = {};
#pragma unroll
    for (int p = 0; p < 4; ++p) {
      a[2 * p]     = (_Float16)fmaxf(fmaf(d, dw0[p], dv0[p]), 0.f);
      a[2 * p + 1] = (_Float16)fmaxf(fmaf(d, dw1[p], dv1[p]), 0.f);
    }
    // accumulator preloaded with P_i + P_j + (ub1 + db2@uW1d)
    v8f c0 = {}, c1 = {};
#pragma unroll
    for (int r = 0; r < 8; ++r) {
      const float* Pj = P + (b * Nn + j0 + r + 8 * hf) * 64 + 32;
      c0[r] = pi0 + Pj[ln] + cc0;
      c1[r] = pi1 + Pj[16 + ln] + cc1;
    }
    c0 = wmma_f16(a, bm0, c0);   // pre_h[:, 0:16]
    c1 = wmma_f16(a, bm1, c1);   // pre_h[:, 16:32]
    // layernorm over 32 feats per row (features lie across lanes) + relu
#pragma unroll
    for (int r = 0; r < 8; ++r) {
      float s = c0[r] + c1[r];
      float q = c0[r] * c0[r] + c1[r] * c1[r];
#pragma unroll
      for (int off = 1; off < 16; off <<= 1) {  // stays within 16-lane half
        s += __shfl_xor(s, off);
        q += __shfl_xor(q, off);
      }
      float mu = s * (1.f / 32.f);
      float var = q * (1.f / 32.f) - mu * mu;
      float rs = rsqrtf(var + LN_EPS);
      float y0 = fmaxf((c0[r] - mu) * rs * g0v + be0, 0.f);
      float y1 = fmaxf((c1[r] - mu) * rs * g1v + be1, 0.f);
      int M = r + 8 * hf;
      lw[M * 32 + ln]      = (_Float16)y0;
      lw[M * 32 + 16 + ln] = (_Float16)y1;
    }
    asm volatile("s_wait_dscnt 0" ::: "memory");  // C-layout -> A-layout
    const unsigned* lwu = (const unsigned*)lw;
    v16h a2 = {};
#pragma unroll
    for (int p = 0; p < 8; ++p) {
      HU t; t.u = lwu[(ln * 32 + kb(p, hf)) >> 1];  // one ds_load_b32 / pair
      a2[2 * p] = t.h[0];
      a2[2 * p + 1] = t.h[1];
    }
    v8f d2 = {};
#pragma unroll
    for (int r = 0; r < 8; ++r) d2[r] = u2;
    d2 = wmma_f16(a2, b2, d2);   // edge_features tile [16 j x 16 e]
    float* out = ef + ((size_t)(b * Nn + i) * Nn + j0) * Ee;
#pragma unroll
    for (int r = 0; r < 8; ++r) {
      float v = d2[r] * scale;
      out[(r + 8 * hf) * Ee + ln] = v;
      poolacc += v;
    }
  }
  poolacc += __shfl_xor(poolacc, 16);
  if (hf == 0) pbuf[wv * 16 + ln] = poolacc;
  __syncthreads();
  if (threadIdx.x < 16)
    pooled[(b * Nn + i) * Ee + ln] =
        pbuf[ln] + pbuf[16 + ln] + pbuf[32 + ln] + pbuf[48 + ln];
}

// ---- node MLP: relu(mean_pool @ nW1 + nb1) @ nW2 + nb2 ---------------------
__global__ void kmsg_kernel(const float* __restrict__ pooled,
                            const float* __restrict__ nb1,
                            const float* __restrict__ nb2,
                            const _Float16* __restrict__ nw1f,
                            const _Float16* __restrict__ nw2f,
                            float* __restrict__ msg) {
  __shared__ _Float16 mid[16 * 64];
  int lane = threadIdx.x & 31, hf = lane >> 4, ln = lane & 15;
  int node0 = blockIdx.x * 16;
  // A: pooled mean [16x16] padded to K=32; 8 consecutive floats per lane
  const float4* p4 = (const float4*)(pooled + (size_t)(node0 + ln) * Ee);
  float4 q0 = p4[(hf * 8) >> 2], q1 = p4[((hf * 8) >> 2) + 1];
  v16h a = {};
  a[0] = (_Float16)(q0.x * (1.f / Nn)); a[1] = (_Float16)(q0.y * (1.f / Nn));
  a[2] = (_Float16)(q0.z * (1.f / Nn)); a[3] = (_Float16)(q0.w * (1.f / Nn));
  a[4] = (_Float16)(q1.x * (1.f / Nn)); a[5] = (_Float16)(q1.y * (1.f / Nn));
  a[6] = (_Float16)(q1.z * (1.f / Nn)); a[7] = (_Float16)(q1.w * (1.f / Nn));
#pragma unroll
  for (int t = 0; t < 4; ++t) {
    v16h bf = *(const v16h*)(nw1f + (t * 32 + lane) * 16);
    v8f c = {};
    float bias = nb1[t * 16 + ln];
#pragma unroll
    for (int r = 0; r < 8; ++r) c[r] = bias;
    c = wmma_f16(a, bf, c);
#pragma unroll
    for (int r = 0; r < 8; ++r)
      mid[(r + 8 * hf) * 64 + t * 16 + ln] = (_Float16)fmaxf(c[r], 0.f);
  }
  asm volatile("s_wait_dscnt 0" ::: "memory");
  const unsigned* midu = (const unsigned*)mid;
#pragma unroll
  for (int t2 = 0; t2 < 8; ++t2) {
    v8f c = {};
    float bias = nb2[t2 * 16 + ln];
#pragma unroll
    for (int r = 0; r < 8; ++r) c[r] = bias;
#pragma unroll
    for (int kc = 0; kc < 2; ++kc) {
      v16h a2 = {};
#pragma unroll
      for (int p = 0; p < 8; ++p) {
        HU t; t.u = midu[(ln * 64 + kc * 32 + kb(p, hf)) >> 1];
        a2[2 * p] = t.h[0];
        a2[2 * p + 1] = t.h[1];
      }
      v16h bf = *(const v16h*)(nw2f + ((t2 * 2 + kc) * 32 + lane) * 16);
      c = wmma_f16(a2, bf, c);
    }
#pragma unroll
    for (int r = 0; r < 8; ++r)
      msg[(node0 + r + 8 * hf) * Hh + t2 * 16 + ln] = c[r];
  }
}

extern "C" void kernel_launch(void* const* d_in, const int* in_sizes, int n_in,
                              void* d_out, int out_size, void* d_ws,
                              size_t ws_size, hipStream_t stream) {
  (void)in_sizes; (void)n_in; (void)out_size; (void)ws_size;
  const float* emb  = (const float*)d_in[0];
  const float* dist = (const float*)d_in[1];
  const int*   vis  = (const int*)d_in[2];
  const float* dW1  = (const float*)d_in[3];
  const float* db1  = (const float*)d_in[4];
  const float* dW2  = (const float*)d_in[5];
  const float* db2  = (const float*)d_in[6];
  const float* uW1  = (const float*)d_in[7];
  const float* ub1  = (const float*)d_in[8];
  const float* lng  = (const float*)d_in[9];
  const float* lnb  = (const float*)d_in[10];
  const float* uW2  = (const float*)d_in[11];
  const float* ub2  = (const float*)d_in[12];
  const float* nW1  = (const float*)d_in[13];
  const float* nb1  = (const float*)d_in[14];
  const float* nW2  = (const float*)d_in[15];
  const float* nb2  = (const float*)d_in[16];

  float* ef  = (float*)d_out;                   // [16,256,256,16]
  float* msg = ef + (size_t)Bb * Nn * Nn * Ee;  // [16,256,128]

  char* ws = (char*)d_ws;
  float*    P        = (float*)(ws + OFF_P);
  float*    pooled   = (float*)(ws + OFF_POOL);
  float*    colconst = (float*)(ws + OFF_CC);
  _Float16* bmf      = (_Float16*)(ws + OFF_BM);
  _Float16* b2f      = (_Float16*)(ws + OFF_B2);
  _Float16* uw1f     = (_Float16*)(ws + OFF_UW1F);
  _Float16* nw1f     = (_Float16*)(ws + OFF_NW1F);
  _Float16* nw2f     = (_Float16*)(ws + OFF_NW2F);

  kconst_kernel<<<1, 32, 0, stream>>>(dW2, db2, uW1, ub1, uW2, nW1, nW2,
                                      colconst, bmf, b2f, uw1f, nw1f, nw2f);
  kP_kernel<<<Bb * Nn / 16, 32, 0, stream>>>(emb, uw1f, P);
  kedge_kernel<<<Bb * Nn, 128, 0, stream>>>(dist, vis, dW1, db1, lng, lnb, ub2,
                                            P, bmf, b2f, colconst, ef, pooled);
  kmsg_kernel<<<Bb * Nn / 16, 32, 0, stream>>>(pooled, nb1, nb2, nw1f, nw2f,
                                               msg);
}
